// RSSMCore_44367012167759
// MI455X (gfx1250) — matrix operand order repaired and checked by reading, not compiled, not run on hardware
//
#include <hip/hip_runtime.h>
#include <math.h>

// ---------------------------------------------------------------------------
// RSSM forward for MI455X (gfx1250, wave32).
// Strategy: weights converted fp32->bf16 once into d_ws; all large GEMMs via
// v_wmma_f32_16x16x32_bf16 (f32 accumulate); activations fp32, staged to LDS
// as bf16 tiles per K-chunk. Sequential T=64 scan = 3 kernels/step + one
// batched prior kernel at the end.
// ---------------------------------------------------------------------------

#define T_STEPS 64
#define BATCH   256
#define EMBED_D 1536
#define ACT_D   10
#define DETER_D 2048
#define STOCH_D 64
#define HID_D   1024
#define LN_EPS  1e-3f

typedef __bf16 bf16_t;
typedef __attribute__((ext_vector_type(16))) __bf16 v16bf;
typedef __attribute__((ext_vector_type(8)))  float  v8f;

__device__ __forceinline__ v8f wmma_bf16(v16bf a, v16bf b, v8f c) {
  // D = A(16x32) * B(32x16) + C, f32 accumulate
  return __builtin_amdgcn_wmma_f32_16x16x32_bf16(
      false, a, false, b, (short)0, c, false, false);
}

__device__ __forceinline__ v8f vzero8() {
  v8f z = {0.f, 0.f, 0.f, 0.f, 0.f, 0.f, 0.f, 0.f};
  return z;
}

// B fragment: K=32 x N=16 bf16 sourced from row-major weight W[O][K].
// B[k][n] = W[o0+n][k0+k]. Lane n reads two 8-bf16 chunks: K-base..+7 and
// K-base+16..+23, K-base = (l<16)?0:8 (same K order as the A fragment).
__device__ __forceinline__ v16bf load_b_frag(const bf16_t* __restrict__ W,
                                             int ldk, int o0, int k0) {
  typedef __attribute__((ext_vector_type(8))) __bf16 v8bf;
  int l = threadIdx.x & 31;
  int n = l & 15;
  int koff = k0 + ((l < 16) ? 0 : 8);
  const bf16_t* p = W + (size_t)(o0 + n) * ldk + koff;
  v8bf lo = *(const v8bf*)p;
  v8bf hi = *(const v8bf*)(p + 16);
  return __builtin_shufflevector(lo, hi, 0, 1, 2, 3, 4, 5, 6, 7, 8, 9, 10, 11, 12, 13, 14, 15);
}

// A fragment: M=16 x K=32 bf16 gathered from LDS rows (stride in bf16, even).
// Lane l: row = l&15, K-base = (l<16)?0:8; VGPR v0..3 -> K pairs base+2v,
// v4..7 -> K pairs base+16+2v (per ISA 16-bit A-matrix table).
__device__ __forceinline__ v16bf load_a_frag_lds(const bf16_t* a0, int stride) {
  int l = threadIdx.x & 31;
  int row = l & 15;
  int kb2 = (l < 16) ? 0 : 4;  // (K-base)/2 in packed words
  union { unsigned u[8]; v16bf v; } r;
  const unsigned* base = (const unsigned*)(a0 + row * stride);
#pragma unroll
  for (int v = 0; v < 4; ++v) r.u[v] = base[kb2 + v];
#pragma unroll
  for (int v = 0; v < 4; ++v) r.u[4 + v] = base[kb2 + 8 + v];
  return r.v;
}

// Cooperatively stage a 16x32 fp32 tile -> bf16 LDS (row stride 34 = conflict
// free: 68B/row -> 17 banks, coprime with 64). 256 threads, 1 pair each.
__device__ __forceinline__ void stage_a(bf16_t* s, const float* __restrict__ A,
                                        int lda, int row0, int k0) {
  int t = threadIdx.x;
  int r = t >> 4;
  int kp = (t & 15) << 1;
  const float* p = A + (size_t)(row0 + r) * lda + k0 + kp;
  union { bf16_t h[2]; unsigned u; } pk;
  pk.h[0] = (bf16_t)p[0];
  pk.h[1] = (bf16_t)p[1];
  *(unsigned*)(s + r * 34 + kp) = pk.u;
}

__device__ __forceinline__ float sigmoidf_(float x) {
  return 1.f / (1.f + __expf(-x));
}
__device__ __forceinline__ float eluf_(float x) {
  return x > 0.f ? x : (__expf(x) - 1.f);
}

// ---------------------------------------------------------------------------
// fp32 -> bf16 weight conversion
// ---------------------------------------------------------------------------
__global__ __launch_bounds__(256) void cvt_f32_bf16(const float* __restrict__ src,
                                                    bf16_t* __restrict__ dst, int n) {
  int i = blockIdx.x * 256 + threadIdx.x;
  if (i < n) dst[i] = (bf16_t)src[i];
}

// ---------------------------------------------------------------------------
// za = ELU(LN(concat(z @ z_mlp_w.T + b, a @ a_mlp_w.T)))   [B,1024]
// one block per batch row; tiny K (64 / 10) -> scalar VALU is right here.
// ---------------------------------------------------------------------------
__global__ __launch_bounds__(256) void za_kernel(
    const float* __restrict__ z_prev,  // [B,64]
    const float* __restrict__ act,     // [B,10]
    const float* __restrict__ zw,      // [512,64]
    const float* __restrict__ zb,      // [512]
    const float* __restrict__ aw,      // [512,10]
    const float* __restrict__ lns, const float* __restrict__ lnb,
    float* __restrict__ za) {          // [B,1024]
  __shared__ float xs[1024];
  __shared__ float red[256];
  int b = blockIdx.x, tid = threadIdx.x;
  float s = 0.f, q = 0.f;
#pragma unroll
  for (int i = 0; i < 4; ++i) {
    int j = tid + 256 * i;
    float v;
    if (j < 512) {
      v = zb[j];
      const float* zr = z_prev + (size_t)b * STOCH_D;
      const float* wr = zw + (size_t)j * STOCH_D;
      for (int k = 0; k < STOCH_D; ++k) v += zr[k] * wr[k];
    } else {
      int jj = j - 512;
      v = 0.f;
      const float* ar = act + (size_t)b * ACT_D;
      const float* wr = aw + (size_t)jj * ACT_D;
      for (int k = 0; k < ACT_D; ++k) v += ar[k] * wr[k];
    }
    xs[j] = v;
    s += v; q += v * v;
  }
  red[tid] = s; __syncthreads();
  for (int o = 128; o > 0; o >>= 1) { if (tid < o) red[tid] += red[tid + o]; __syncthreads(); }
  float mean = red[0] * (1.f / 1024.f);
  __syncthreads();
  red[tid] = q; __syncthreads();
  for (int o = 128; o > 0; o >>= 1) { if (tid < o) red[tid] += red[tid + o]; __syncthreads(); }
  float var = red[0] * (1.f / 1024.f) - mean * mean;
  float ri = rsqrtf(var + LN_EPS);
#pragma unroll
  for (int i = 0; i < 4; ++i) {
    int j = tid + 256 * i;
    float v = (xs[j] - mean) * ri * lns[j] + lnb[j];
    za[(size_t)b * HID_D + j] = eluf_(v);
  }
}

// ---------------------------------------------------------------------------
// GRU: h1 = (1-u)*c + u*h.  One wave = one 16x16 tile of h1, carrying six
// WMMA accumulators (r/u/c for ih and hh). Block = 16 rows x 128 cols.
// ---------------------------------------------------------------------------
__global__ __launch_bounds__(256) void gru_kernel(
    const float* __restrict__ za,      // [B,1024]
    const float* __restrict__ h_prev, int ldh,
    const bf16_t* __restrict__ wih,    // [6144,1024] bf16
    const bf16_t* __restrict__ whh,    // [6144,2048] bf16
    const float* __restrict__ bih, const float* __restrict__ bhh,
    float* __restrict__ h_out, int ldo) {
  __shared__ bf16_t a_stage[16 * 34];
  int bm = blockIdx.x * 16;
  int wave = threadIdx.x >> 5, lane = threadIdx.x & 31;
  int dn = blockIdx.y * 128 + wave * 16;

  v8f aRi = vzero8(), aUi = vzero8(), aCi = vzero8();
  v8f aRh = vzero8(), aUh = vzero8(), aCh = vzero8();

  // phase 1: gi = za @ wih.T    (K = 1024)
  for (int k0 = 0; k0 < HID_D; k0 += 32) {
    __syncthreads();
    stage_a(a_stage, za, HID_D, bm, k0);
    __syncthreads();
    v16bf a = load_a_frag_lds(a_stage, 34);
    v16bf bR = load_b_frag(wih, HID_D, 0 * DETER_D + dn, k0);
    v16bf bU = load_b_frag(wih, HID_D, 1 * DETER_D + dn, k0);
    v16bf bC = load_b_frag(wih, HID_D, 2 * DETER_D + dn, k0);
    aRi = wmma_bf16(a, bR, aRi);
    aUi = wmma_bf16(a, bU, aUi);
    aCi = wmma_bf16(a, bC, aCi);
  }
  // phase 2: gh = h @ whh.T     (K = 2048)
  for (int k0 = 0; k0 < DETER_D; k0 += 32) {
    __syncthreads();
    stage_a(a_stage, h_prev, ldh, bm, k0);
    __syncthreads();
    v16bf a = load_a_frag_lds(a_stage, 34);
    v16bf bR = load_b_frag(whh, DETER_D, 0 * DETER_D + dn, k0);
    v16bf bU = load_b_frag(whh, DETER_D, 1 * DETER_D + dn, k0);
    v16bf bC = load_b_frag(whh, DETER_D, 2 * DETER_D + dn, k0);
    aRh = wmma_bf16(a, bR, aRh);
    aUh = wmma_bf16(a, bU, aUh);
    aCh = wmma_bf16(a, bC, aCh);
  }
  int M0 = (lane < 16) ? 0 : 8;
  int N = lane & 15;
  int d = dn + N;
  float biR = bih[d],           bhR = bhh[d];
  float biU = bih[DETER_D + d], bhU = bhh[DETER_D + d];
  float biC = bih[2 * DETER_D + d], bhC = bhh[2 * DETER_D + d];
#pragma unroll
  for (int v = 0; v < 8; ++v) {
    int brow = bm + M0 + v;
    float r = sigmoidf_((aRi[v] + biR) + (aRh[v] + bhR));
    float u = sigmoidf_((aUi[v] + biU) + (aUh[v] + bhU));
    float c = tanhf((aCi[v] + biC) + r * (aCh[v] + bhC));
    float hp = h_prev[(size_t)brow * ldh + d];
    h_out[(size_t)brow * ldo + d] = (1.f - u) * c + u * hp;
  }
}

// ---------------------------------------------------------------------------
// Posterior (fused): p = ELU(LN(h1@Wh.T + bh + e@We.T + be));
// post = p@Wo.T + bo; mean/std split; z1 = mean + std*noise.
// Block = 16 batch rows. 8 waves x 8 col-tiles cover hidden(1024); then
// LN+ELU in LDS (bf16), then 8 waves x 1 tile cover out(128).
// ---------------------------------------------------------------------------
__global__ __launch_bounds__(256) void post_kernel(
    const float* __restrict__ h1, int ldh1,      // features[t], ld 2112
    const float* __restrict__ emb,               // [B,1536]
    const bf16_t* __restrict__ wh,               // [1024,2048]
    const bf16_t* __restrict__ we,               // [1024,1536]
    const float* __restrict__ bh, const float* __restrict__ be,
    const float* __restrict__ lns, const float* __restrict__ lnb,
    const bf16_t* __restrict__ wo,               // [128,1024]
    const float* __restrict__ bo,                // [128]
    const float* __restrict__ noise_t,           // [B,64]
    float* __restrict__ posts_t,                 // [B,128]
    float* __restrict__ z_out,                   // [B,64]
    float* __restrict__ feat_t) {                // features[t] (z tail)
  __shared__ bf16_t a_stage[16 * 34];
  __shared__ bf16_t hid[16][1042];               // padded: 521 words/row
  __shared__ float post_s[16][129];
  __shared__ float reds[256], redq[256];
  __shared__ float redm[16], redr[16];
  int bm = blockIdx.x * 16;
  int wave = threadIdx.x >> 5, lane = threadIdx.x & 31;
  int M0 = (lane < 16) ? 0 : 8;
  int N = lane & 15;

  v8f acc[8];
#pragma unroll
  for (int i = 0; i < 8; ++i) acc[i] = vzero8();

  for (int k0 = 0; k0 < DETER_D; k0 += 32) {   // h path
    __syncthreads();
    stage_a(a_stage, h1, ldh1, bm, k0);
    __syncthreads();
    v16bf a = load_a_frag_lds(a_stage, 34);
#pragma unroll
    for (int i = 0; i < 8; ++i)
      acc[i] = wmma_bf16(a, load_b_frag(wh, DETER_D, (i * 8 + wave) * 16, k0), acc[i]);
  }
  for (int k0 = 0; k0 < EMBED_D; k0 += 32) {   // e path
    __syncthreads();
    stage_a(a_stage, emb, EMBED_D, bm, k0);
    __syncthreads();
    v16bf a = load_a_frag_lds(a_stage, 34);
#pragma unroll
    for (int i = 0; i < 8; ++i)
      acc[i] = wmma_bf16(a, load_b_frag(we, EMBED_D, (i * 8 + wave) * 16, k0), acc[i]);
  }
  __syncthreads();
#pragma unroll
  for (int i = 0; i < 8; ++i) {
    int col = (i * 8 + wave) * 16 + N;
    float bias = bh[col] + be[col];
#pragma unroll
    for (int v = 0; v < 8; ++v) hid[M0 + v][col] = (bf16_t)(acc[i][v] + bias);
  }
  __syncthreads();

  // LayerNorm + ELU, 16 threads per row
  int row = threadIdx.x >> 4, l16 = threadIdx.x & 15;
  float s = 0.f, q = 0.f;
  for (int i = 0; i < 64; ++i) {
    float v = (float)hid[row][l16 + 16 * i];
    s += v; q += v * v;
  }
  reds[threadIdx.x] = s; redq[threadIdx.x] = q;
  __syncthreads();
  if (l16 == 0) {
    float S = 0.f, Q = 0.f;
    for (int i = 0; i < 16; ++i) { S += reds[row * 16 + i]; Q += redq[row * 16 + i]; }
    float mu = S * (1.f / 1024.f);
    redm[row] = mu;
    redr[row] = rsqrtf(Q * (1.f / 1024.f) - mu * mu + LN_EPS);
  }
  __syncthreads();
  float mu = redm[row], ri = redr[row];
  for (int i = 0; i < 64; ++i) {
    int j = l16 + 16 * i;
    float v = ((float)hid[row][j] - mu) * ri * lns[j] + lnb[j];
    hid[row][j] = (bf16_t)eluf_(v);
  }
  __syncthreads();

  // out GEMM: wave w -> cols [w*16, w*16+16), K = 1024
  v8f o = vzero8();
  for (int k0 = 0; k0 < HID_D; k0 += 32) {
    v16bf a = load_a_frag_lds(&hid[0][0] + k0, 1042);
    o = wmma_bf16(a, load_b_frag(wo, HID_D, wave * 16, k0), o);
  }
  {
    int col = wave * 16 + N;
    float bias = bo[col];
#pragma unroll
    for (int v = 0; v < 8; ++v) {
      float val = o[v] + bias;
      post_s[M0 + v][col] = val;
      posts_t[(size_t)(bm + M0 + v) * 128 + col] = val;
    }
  }
  __syncthreads();

  // z1 = mean + (2*sigmoid(sraw)+0.1)*noise
#pragma unroll
  for (int i = 0; i < 4; ++i) {
    int idx = threadIdx.x + 256 * i;
    int r = idx >> 6, j = idx & 63;
    float mean = post_s[r][j];
    float std = 2.f * sigmoidf_(post_s[r][j + 64]) + 0.1f;
    float z1 = mean + std * noise_t[(size_t)(bm + r) * STOCH_D + j];
    z_out[(size_t)(bm + r) * STOCH_D + j] = z1;
    feat_t[(size_t)(bm + r) * (DETER_D + STOCH_D) + DETER_D + j] = z1;
  }
}

// ---------------------------------------------------------------------------
// Prior (batched over T*B = 16384 rows): same fused structure, h path only.
// ---------------------------------------------------------------------------
__global__ __launch_bounds__(256) void prior_kernel(
    const float* __restrict__ feats,             // [T*B, 2112], h in cols 0..2047
    const bf16_t* __restrict__ wh,               // [1024,2048]
    const float* __restrict__ bh,
    const float* __restrict__ lns, const float* __restrict__ lnb,
    const bf16_t* __restrict__ wo,               // [128,1024]
    const float* __restrict__ bo,
    float* __restrict__ priors) {                // [T*B,128]
  __shared__ bf16_t a_stage[16 * 34];
  __shared__ bf16_t hid[16][1042];
  __shared__ float reds[256], redq[256];
  __shared__ float redm[16], redr[16];
  int bm = blockIdx.x * 16;
  int wave = threadIdx.x >> 5, lane = threadIdx.x & 31;
  int M0 = (lane < 16) ? 0 : 8;
  int N = lane & 15;

  v8f acc[8];
#pragma unroll
  for (int i = 0; i < 8; ++i) acc[i] = vzero8();

  for (int k0 = 0; k0 < DETER_D; k0 += 32) {
    __syncthreads();
    stage_a(a_stage, feats, DETER_D + STOCH_D, bm, k0);
    __syncthreads();
    v16bf a = load_a_frag_lds(a_stage, 34);
#pragma unroll
    for (int i = 0; i < 8; ++i)
      acc[i] = wmma_bf16(a, load_b_frag(wh, DETER_D, (i * 8 + wave) * 16, k0), acc[i]);
  }
  __syncthreads();
#pragma unroll
  for (int i = 0; i < 8; ++i) {
    int col = (i * 8 + wave) * 16 + N;
    float bias = bh[col];
#pragma unroll
    for (int v = 0; v < 8; ++v) hid[M0 + v][col] = (bf16_t)(acc[i][v] + bias);
  }
  __syncthreads();

  int row = threadIdx.x >> 4, l16 = threadIdx.x & 15;
  float s = 0.f, q = 0.f;
  for (int i = 0; i < 64; ++i) {
    float v = (float)hid[row][l16 + 16 * i];
    s += v; q += v * v;
  }
  reds[threadIdx.x] = s; redq[threadIdx.x] = q;
  __syncthreads();
  if (l16 == 0) {
    float S = 0.f, Q = 0.f;
    for (int i = 0; i < 16; ++i) { S += reds[row * 16 + i]; Q += redq[row * 16 + i]; }
    float mu = S * (1.f / 1024.f);
    redm[row] = mu;
    redr[row] = rsqrtf(Q * (1.f / 1024.f) - mu * mu + LN_EPS);
  }
  __syncthreads();
  float mu = redm[row], ri = redr[row];
  for (int i = 0; i < 64; ++i) {
    int j = l16 + 16 * i;
    float v = ((float)hid[row][j] - mu) * ri * lns[j] + lnb[j];
    hid[row][j] = (bf16_t)eluf_(v);
  }
  __syncthreads();

  v8f o = vzero8();
  for (int k0 = 0; k0 < HID_D; k0 += 32) {
    v16bf a = load_a_frag_lds(&hid[0][0] + k0, 1042);
    o = wmma_bf16(a, load_b_frag(wo, HID_D, wave * 16, k0), o);
  }
  int col = wave * 16 + N;
  float bias = bo[col];
#pragma unroll
  for (int v = 0; v < 8; ++v)
    priors[(size_t)(bm + M0 + v) * 128 + col] = o[v] + bias;
}

// ---------------------------------------------------------------------------
// Host launch
// ---------------------------------------------------------------------------
extern "C" void kernel_launch(void* const* d_in, const int* in_sizes, int n_in,
                              void* d_out, int out_size, void* d_ws, size_t ws_size,
                              hipStream_t stream) {
  (void)in_sizes; (void)n_in; (void)out_size; (void)ws_size;
  const float* embeds  = (const float*)d_in[0];   // [T,B,1536]
  const float* actions = (const float*)d_in[1];   // [T,B,10]
  const float* h0      = (const float*)d_in[2];   // [B,2048]
  const float* z0      = (const float*)d_in[3];   // [B,64]
  const float* noise   = (const float*)d_in[4];   // [T,B,64]
  const float* z_mlp_w = (const float*)d_in[5];
  const float* z_mlp_b = (const float*)d_in[6];
  const float* a_mlp_w = (const float*)d_in[7];
  const float* ln_za_s = (const float*)d_in[8];
  const float* ln_za_b = (const float*)d_in[9];
  const float* gru_wih = (const float*)d_in[10];  // [6144,1024]
  const float* gru_whh = (const float*)d_in[11];  // [6144,2048]
  const float* gru_bih = (const float*)d_in[12];
  const float* gru_bhh = (const float*)d_in[13];
  const float* post_h_w = (const float*)d_in[14]; // [1024,2048]
  const float* post_h_b = (const float*)d_in[15];
  const float* post_e_w = (const float*)d_in[16]; // [1024,1536]
  const float* post_e_b = (const float*)d_in[17];
  const float* ln_post_s = (const float*)d_in[18];
  const float* ln_post_b = (const float*)d_in[19];
  const float* post_w = (const float*)d_in[20];   // [128,1024]
  const float* post_b = (const float*)d_in[21];
  const float* prior_h_w = (const float*)d_in[22];// [1024,2048]
  const float* prior_h_b = (const float*)d_in[23];
  const float* ln_prior_s = (const float*)d_in[24];
  const float* ln_prior_b = (const float*)d_in[25];
  const float* prior_w = (const float*)d_in[26];  // [128,1024]
  const float* prior_b = (const float*)d_in[27];

  // output layout (flat, return order)
  float* priors = (float*)d_out;
  float* posts  = priors + (size_t)T_STEPS * BATCH * 128;
  float* zsamp  = posts  + (size_t)T_STEPS * BATCH * 128;
  float* feats  = zsamp  + (size_t)T_STEPS * BATCH * STOCH_D;
  const int LDF = DETER_D + STOCH_D;  // 2112

  // workspace layout
  char* ws = (char*)d_ws;
  size_t off = 0;
  auto take = [&](size_t bytes) { size_t o = off; off = (off + bytes + 255) & ~(size_t)255; return o; };
  bf16_t* wih_bf   = (bf16_t*)(ws + take((size_t)6144 * 1024 * 2));
  bf16_t* whh_bf   = (bf16_t*)(ws + take((size_t)6144 * 2048 * 2));
  bf16_t* posth_bf = (bf16_t*)(ws + take((size_t)1024 * 2048 * 2));
  bf16_t* poste_bf = (bf16_t*)(ws + take((size_t)1024 * 1536 * 2));
  bf16_t* priorh_bf= (bf16_t*)(ws + take((size_t)1024 * 2048 * 2));
  bf16_t* postw_bf = (bf16_t*)(ws + take((size_t)128 * 1024 * 2));
  bf16_t* priorw_bf= (bf16_t*)(ws + take((size_t)128 * 1024 * 2));
  float*  za_buf   = (float*)(ws + take((size_t)BATCH * HID_D * 4));

  auto cvt = [&](const float* src, bf16_t* dst, int n) {
    cvt_f32_bf16<<<(n + 255) / 256, 256, 0, stream>>>(src, dst, n);
  };
  cvt(gru_wih,   wih_bf,   6144 * 1024);
  cvt(gru_whh,   whh_bf,   6144 * 2048);
  cvt(post_h_w,  posth_bf, 1024 * 2048);
  cvt(post_e_w,  poste_bf, 1024 * 1536);
  cvt(prior_h_w, priorh_bf,1024 * 2048);
  cvt(post_w,    postw_bf, 128 * 1024);
  cvt(prior_w,   priorw_bf,128 * 1024);

  for (int t = 0; t < T_STEPS; ++t) {
    const float* zprev = (t == 0) ? z0 : (zsamp + (size_t)(t - 1) * BATCH * STOCH_D);
    const float* hprev = (t == 0) ? h0 : (feats + (size_t)(t - 1) * BATCH * LDF);
    int ldh = (t == 0) ? DETER_D : LDF;
    float* feat_t = feats + (size_t)t * BATCH * LDF;

    za_kernel<<<BATCH, 256, 0, stream>>>(
        zprev, actions + (size_t)t * BATCH * ACT_D,
        z_mlp_w, z_mlp_b, a_mlp_w, ln_za_s, ln_za_b, za_buf);

    gru_kernel<<<dim3(BATCH / 16, DETER_D / 128), 256, 0, stream>>>(
        za_buf, hprev, ldh, wih_bf, whh_bf, gru_bih, gru_bhh, feat_t, LDF);

    post_kernel<<<BATCH / 16, 256, 0, stream>>>(
        feat_t, LDF, embeds + (size_t)t * BATCH * EMBED_D,
        posth_bf, poste_bf, post_h_b, post_e_b, ln_post_s, ln_post_b,
        postw_bf, post_b, noise + (size_t)t * BATCH * STOCH_D,
        posts + (size_t)t * BATCH * 128, zsamp + (size_t)t * BATCH * STOCH_D,
        feat_t);
  }

  prior_kernel<<<(T_STEPS * BATCH) / 16, 256, 0, stream>>>(
      feats, priorh_bf, prior_h_b, ln_prior_s, ln_prior_b,
      priorw_bf, prior_b, priors);
}
